// MHSA_28793460753113
// MI455X (gfx1250) — compile-verified, matrix operand order checked by reading, not run on hardware
//
#include <hip/hip_runtime.h>
#include <math.h>

#define S_LEN 4096
#define EMB   768
#define NH    12
#define HD    64
#define BATCH 8
#define MTOT  (BATCH * S_LEN)   // 32768
#define NQKV  (3 * EMB)         // 2304
#define SCALE 0.036084391824351615f  // 768^-0.5 (reference uses emb**-0.5)

#define BM 128
#define BN 128
#define BK 64
#define KTILES (EMB / BK)
#define LSTR 72   // LDS row stride in halfs: 64 data + 8 pad (144B, 16B aligned)

typedef __attribute__((ext_vector_type(16))) __bf16        v16bf;
typedef __attribute__((ext_vector_type(8)))  float         v8f;
typedef __attribute__((ext_vector_type(4)))  unsigned int  u32x4;
typedef __attribute__((ext_vector_type(2)))  unsigned int  u32x2;
typedef __attribute__((ext_vector_type(4)))  float         f32x4;

union FragBF { v16bf v; u32x4 q[2]; };

static __device__ __forceinline__ unsigned short f2bf(float x) {
    __bf16 h = (__bf16)x;                       // RNE convert (v_cvt)
    return __builtin_bit_cast(unsigned short, h);
}
static __device__ __forceinline__ float bf2f(unsigned short h) {
    return (float)__builtin_bit_cast(__bf16, h);
}
static __device__ __forceinline__ unsigned pack2(float a, float b) {
    return (unsigned)f2bf(a) | ((unsigned)f2bf(b) << 16);
}

// ---- CDNA5 async DMA: global -> LDS, tracked by ASYNCcnt ------------------
static __device__ __forceinline__ void async_b128(unsigned short* lds,
                                                  const unsigned short* g) {
    unsigned       ldsOff = (unsigned)(size_t)lds;            // addr[31:0] = LDS offset
    unsigned long long ga = (unsigned long long)(size_t)g;
    asm volatile("global_load_async_to_lds_b128 %0, %1, off"
                 :: "v"(ldsOff), "v"(ga) : "memory");
}
static __device__ __forceinline__ void wait_async0() {
    asm volatile("s_wait_asynccnt 0x0" ::: "memory");
}

// ---------------------------------------------------------------------------
// Kernel 0: elementwise fp32 -> bf16 (prep so GEMMs can DMA bf16 into LDS)
// ---------------------------------------------------------------------------
__global__ __launch_bounds__(256)
void cvt_f32_bf16_kernel(const float* __restrict__ src,
                         unsigned short* __restrict__ dst, int n4)
{
    int i = blockIdx.x * 256 + threadIdx.x;
    if (i < n4) {
        f32x4 s = *(const f32x4*)(src + (size_t)i * 4);
        u32x2 p; p.x = pack2(s.x, s.y); p.y = pack2(s.z, s.w);
        *(u32x2*)(dst + (size_t)i * 4) = p;
    }
}

// ---------------------------------------------------------------------------
// Shared GEMM core: issue one 128x64 A tile + 128x64 B tile via async DMA
// ---------------------------------------------------------------------------
static __device__ __forceinline__ void issue_tile(
    unsigned short* lA, unsigned short* lB,
    const unsigned short* __restrict__ Ab, const unsigned short* __restrict__ Bb,
    int mBase, int nBase, int kb, int tid)
{
    #pragma unroll
    for (int i = 0; i < 4; ++i) {
        int c = tid + i * 256;                  // 1024 chunks of 8 halfs
        int r = c >> 3, cc = c & 7;
        async_b128(&lA[r * LSTR + cc * 8],
                   Ab + (size_t)(mBase + r) * EMB + kb + cc * 8);
    }
    #pragma unroll
    for (int i = 0; i < 4; ++i) {
        int c = tid + i * 256;
        int r = c >> 3, cc = c & 7;
        async_b128(&lB[r * LSTR + cc * 8],
                   Bb + (size_t)(nBase + r) * EMB + kb + cc * 8);
    }
}

// ---------------------------------------------------------------------------
// Kernel 1: QKV projection. qkv[32768,2304] = Xb[32768,768] @ Wqkv_b[2304,768]^T
// Async double-buffered staging; WMMA bf16 f32-accumulate; bf16 output rows.
// ---------------------------------------------------------------------------
__global__ __launch_bounds__(256)
void qkv_gemm_kernel(const unsigned short* __restrict__ Xb,
                     const unsigned short* __restrict__ Wb,
                     unsigned short* __restrict__ QKV)
{
    __shared__ unsigned short lA[2][BM * LSTR];
    __shared__ unsigned short lB[2][BN * LSTR];

    const int tid   = threadIdx.x;
    const int lane  = tid & 31;
    const int wave  = tid >> 5;
    const int waveM = wave >> 2;
    const int waveN = wave & 3;
    const int mBase = blockIdx.y * BM;
    const int nBase = blockIdx.x * BN;

    v8f acc[4][2];
    #pragma unroll
    for (int i = 0; i < 4; ++i)
        #pragma unroll
        for (int j = 0; j < 2; ++j) acc[i][j] = v8f{};

    issue_tile(lA[0], lB[0], Xb, Wb, mBase, nBase, 0, tid);

    for (int kt = 0; kt < KTILES; ++kt) {
        wait_async0();                 // my tile-kt DMAs are in LDS
        __syncthreads();               // everyone's are; prev compute done
        if (kt + 1 < KTILES)
            issue_tile(lA[(kt + 1) & 1], lB[(kt + 1) & 1], Xb, Wb,
                       mBase, nBase, (kt + 1) * BK, tid);

        const unsigned short* cA = lA[kt & 1];
        const unsigned short* cB = lB[kt & 1];
        const int hsel = (lane >> 4) * 8;
        #pragma unroll
        for (int ks = 0; ks < 2; ++ks) {
            FragBF a[4], b[2];
            #pragma unroll
            for (int mi = 0; mi < 4; ++mi) {
                int row = waveM * 64 + mi * 16 + (lane & 15);
                const unsigned short* p = &cA[row * LSTR + ks * 32 + hsel];
                a[mi].q[0] = *(const u32x4*)p;
                a[mi].q[1] = *(const u32x4*)(p + 16);
            }
            #pragma unroll
            for (int ni = 0; ni < 2; ++ni) {
                int row = waveN * 32 + ni * 16 + (lane & 15);
                const unsigned short* p = &cB[row * LSTR + ks * 32 + hsel];
                b[ni].q[0] = *(const u32x4*)p;
                b[ni].q[1] = *(const u32x4*)(p + 16);
            }
            #pragma unroll
            for (int mi = 0; mi < 4; ++mi)
                #pragma unroll
                for (int ni = 0; ni < 2; ++ni)
                    acc[mi][ni] = __builtin_amdgcn_wmma_f32_16x16x32_bf16(
                        false, a[mi].v, false, b[ni].v,
                        (short)0, acc[mi][ni], false, false);
        }
    }

    #pragma unroll
    for (int mi = 0; mi < 4; ++mi)
        #pragma unroll
        for (int ni = 0; ni < 2; ++ni) {
            int n = nBase + waveN * 32 + ni * 16 + (lane & 15);
            #pragma unroll
            for (int r = 0; r < 8; ++r) {
                int m = mBase + waveM * 64 + mi * 16 + r + 8 * (lane >> 4);
                QKV[(size_t)m * NQKV + n] = f2bf(acc[mi][ni][r]);
            }
        }
}

// ---------------------------------------------------------------------------
// Kernel 2: per-position head-mix attention (reference's bsid,bsjd->bsij).
// One wave per position; 12x12 attention over heads, contraction dim 64.
// Output written [B,H,S,D] bf16 so transpose+reshape is a reinterpretation.
// ---------------------------------------------------------------------------
__global__ __launch_bounds__(256)
void headmix_attn_kernel(const unsigned short* __restrict__ QKV,
                         unsigned short* __restrict__ O)
{
    __shared__ unsigned short row[8][NQKV];
    const int lane = threadIdx.x & 31;
    const int wave = threadIdx.x >> 5;
    const int m    = blockIdx.x * 8 + wave;

    const unsigned short* src = QKV + (size_t)m * NQKV;
    #pragma unroll
    for (int i = 0; i < 9; ++i) {
        int c = lane + i * 32;
        *(u32x4*)&row[wave][c * 8] = *(const u32x4*)(src + c * 8);
    }
    __syncthreads();

    if (lane < NH) {
        const unsigned short* q  = &row[wave][lane * HD];
        const unsigned short* kk = &row[wave][EMB];
        const unsigned short* vv = &row[wave][2 * EMB];

        float qf[HD];
        #pragma unroll
        for (int d = 0; d < HD; ++d) qf[d] = bf2f(q[d]);

        float att[NH];
        float mx = -1e30f;
        #pragma unroll
        for (int j = 0; j < NH; ++j) {
            const unsigned short* kj = kk + j * HD;
            float s = 0.f;
            #pragma unroll
            for (int d = 0; d < HD; ++d) s = fmaf(qf[d], bf2f(kj[d]), s);
            s *= SCALE;
            att[j] = s;
            mx = fmaxf(mx, s);
        }
        float sum = 0.f;
        #pragma unroll
        for (int j = 0; j < NH; ++j) { att[j] = __expf(att[j] - mx); sum += att[j]; }
        const float inv = 1.f / sum;

        float o[HD];
        #pragma unroll
        for (int d = 0; d < HD; ++d) o[d] = 0.f;
        #pragma unroll
        for (int j = 0; j < NH; ++j) {
            const float p = att[j] * inv;
            const unsigned short* vj = vv + j * HD;
            #pragma unroll
            for (int d = 0; d < HD; ++d) o[d] = fmaf(p, bf2f(vj[d]), o[d]);
        }

        const int b = m / S_LEN, spos = m % S_LEN;
        unsigned short* dst = O + (((size_t)b * NH + lane) * S_LEN + spos) * HD;
        #pragma unroll
        for (int d0 = 0; d0 < HD; d0 += 8) {
            u32x4 w;
            w.x = pack2(o[d0 + 0], o[d0 + 1]);
            w.y = pack2(o[d0 + 2], o[d0 + 3]);
            w.z = pack2(o[d0 + 4], o[d0 + 5]);
            w.w = pack2(o[d0 + 6], o[d0 + 7]);
            *(u32x4*)(dst + d0) = w;
        }
    }
}

// ---------------------------------------------------------------------------
// Kernel 3: output projection. out[32768,768] = attn_bf16 @ Wp_b[768,768]^T + b
// ---------------------------------------------------------------------------
__global__ __launch_bounds__(256)
void proj_gemm_kernel(const unsigned short* __restrict__ Ab,
                      const unsigned short* __restrict__ Wb,
                      const float* __restrict__ bias,
                      float* __restrict__ OUT)
{
    __shared__ unsigned short lA[2][BM * LSTR];
    __shared__ unsigned short lB[2][BN * LSTR];

    const int tid   = threadIdx.x;
    const int lane  = tid & 31;
    const int wave  = tid >> 5;
    const int waveM = wave >> 2;
    const int waveN = wave & 3;
    const int mBase = blockIdx.y * BM;
    const int nBase = blockIdx.x * BN;

    v8f acc[4][2];
    #pragma unroll
    for (int i = 0; i < 4; ++i)
        #pragma unroll
        for (int j = 0; j < 2; ++j) acc[i][j] = v8f{};

    issue_tile(lA[0], lB[0], Ab, Wb, mBase, nBase, 0, tid);

    for (int kt = 0; kt < KTILES; ++kt) {
        wait_async0();
        __syncthreads();
        if (kt + 1 < KTILES)
            issue_tile(lA[(kt + 1) & 1], lB[(kt + 1) & 1], Ab, Wb,
                       mBase, nBase, (kt + 1) * BK, tid);

        const unsigned short* cA = lA[kt & 1];
        const unsigned short* cB = lB[kt & 1];
        const int hsel = (lane >> 4) * 8;
        #pragma unroll
        for (int ks = 0; ks < 2; ++ks) {
            FragBF a[4], b[2];
            #pragma unroll
            for (int mi = 0; mi < 4; ++mi) {
                int row = waveM * 64 + mi * 16 + (lane & 15);
                const unsigned short* p = &cA[row * LSTR + ks * 32 + hsel];
                a[mi].q[0] = *(const u32x4*)p;
                a[mi].q[1] = *(const u32x4*)(p + 16);
            }
            #pragma unroll
            for (int ni = 0; ni < 2; ++ni) {
                int row = waveN * 32 + ni * 16 + (lane & 15);
                const unsigned short* p = &cB[row * LSTR + ks * 32 + hsel];
                b[ni].q[0] = *(const u32x4*)p;
                b[ni].q[1] = *(const u32x4*)(p + 16);
            }
            #pragma unroll
            for (int mi = 0; mi < 4; ++mi)
                #pragma unroll
                for (int ni = 0; ni < 2; ++ni)
                    acc[mi][ni] = __builtin_amdgcn_wmma_f32_16x16x32_bf16(
                        false, a[mi].v, false, b[ni].v,
                        (short)0, acc[mi][ni], false, false);
        }
    }

    #pragma unroll
    for (int mi = 0; mi < 4; ++mi)
        #pragma unroll
        for (int ni = 0; ni < 2; ++ni) {
            int n = nBase + waveN * 32 + ni * 16 + (lane & 15);
            float bn = bias[n];
            #pragma unroll
            for (int r = 0; r < 8; ++r) {
                int m = mBase + waveM * 64 + mi * 16 + r + 8 * (lane >> 4);
                OUT[(size_t)m * EMB + n] = acc[mi][ni][r] + bn;
            }
        }
}

// ---------------------------------------------------------------------------
extern "C" void kernel_launch(void* const* d_in, const int* in_sizes, int n_in,
                              void* d_out, int out_size, void* d_ws, size_t ws_size,
                              hipStream_t stream)
{
    const float* x      = (const float*)d_in[0];
    const float* w_qkv  = (const float*)d_in[1];
    const float* w_proj = (const float*)d_in[2];
    const float* b_proj = (const float*)d_in[3];
    float*       out    = (float*)d_out;

    const size_t qkvE  = (size_t)MTOT * NQKV;   // 75,497,472 halfs
    const size_t xbE   = (size_t)MTOT * EMB;    // 25,165,824 halfs
    const size_t wqkvE = (size_t)NQKV * EMB;    // 1,769,472 halfs
    const size_t wprjE = (size_t)EMB  * EMB;    //   589,824 halfs

    unsigned short* qkv    = (unsigned short*)d_ws;
    unsigned short* xb     = qkv + qkvE;
    unsigned short* wqkvb  = xb + xbE;
    unsigned short* wprojb = wqkvb + wqkvE;
    unsigned short* attn   = xb;   // xb dead after QKV GEMM -> reuse for attn out

    cvt_f32_bf16_kernel<<<dim3((int)(xbE / 4 / 256)), 256, 0, stream>>>(x, xb, (int)(xbE / 4));
    cvt_f32_bf16_kernel<<<dim3((int)(wqkvE / 4 / 256)), 256, 0, stream>>>(w_qkv, wqkvb, (int)(wqkvE / 4));
    cvt_f32_bf16_kernel<<<dim3((int)(wprjE / 4 / 256)), 256, 0, stream>>>(w_proj, wprojb, (int)(wprjE / 4));

    // grid.x = N-tiles (fastest) so consecutive blocks share A rows -> L2 reuse
    qkv_gemm_kernel<<<dim3(NQKV / BN, MTOT / BM), 256, 0, stream>>>(xb, wqkvb, qkv);
    headmix_attn_kernel<<<dim3(MTOT / 8), 256, 0, stream>>>(qkv, attn);
    proj_gemm_kernel<<<dim3(EMB / BN, MTOT / BM), 256, 0, stream>>>(attn, w_proj ? wprojb : wprojb, b_proj, out);
}
